// GATlayerV1_45105746542631
// MI455X (gfx1250) — matrix operand order repaired and checked by reading, not compiled, not run on hardware
//
#include <hip/hip_runtime.h>
#include <cstdint>

#define IN_F   128
#define BOND_F 64
#define OUT_F  128
#define DEG    8

typedef __attribute__((ext_vector_type(16))) __bf16 v16bf;
typedef __attribute__((ext_vector_type(8)))  float  v8f;

struct Bits256 { uint32_t x[8]; };

__device__ __forceinline__ uint16_t f2bf(float f) {
  uint32_t u = __builtin_bit_cast(uint32_t, f);
  uint32_t r = (u + 0x7FFFu + ((u >> 16) & 1u)) >> 16;   // round-to-nearest-even
  return (uint16_t)r;
}
__device__ __forceinline__ float bf2f(uint16_t h) {
  uint32_t u = ((uint32_t)h) << 16;
  return __builtin_bit_cast(float, u);
}
__device__ __forceinline__ uint32_t pack2(float a, float b) {
  return (uint32_t)f2bf(a) | ((uint32_t)f2bf(b) << 16);
}
__device__ __forceinline__ float lrelu(float x) { return x > 0.f ? x : 0.2f * x; }

// ---- WMMA fragment loaders (CDNA5 wave32 layouts, CDNA5 ISA §7.12.2) ----
// A (16x32 bf16): lane L holds row (L&15); k = (e>>3)*16 + ((L>>4)*8) + (e&7)
// -> two contiguous 16B loads per lane at kb and kb+16 (kb = kbase + 8*(L>=16)).
__device__ __forceinline__ v16bf load_a_bf16(const uint16_t* __restrict__ M, size_t row,
                                             int ld, int kbase, int lane) {
  int kb = kbase + ((lane >> 4) << 3);
  const uint16_t* p = M + row * (size_t)ld + kb;
  uint4 lo = *reinterpret_cast<const uint4*>(p);
  uint4 hi = *reinterpret_cast<const uint4*>(p + 16);
  Bits256 b = {{lo.x, lo.y, lo.z, lo.w, hi.x, hi.y, hi.z, hi.w}};
  return __builtin_bit_cast(v16bf, b);
}
// A fragment converting fp32 source on the fly (only used for the streaming bond matrix).
__device__ __forceinline__ v16bf load_a_f32(const float* __restrict__ M, size_t row,
                                            int ld, int kbase, int lane) {
  int kb = kbase + ((lane >> 4) << 3);
  const float* p = M + row * (size_t)ld + kb;
  float4 f0 = *reinterpret_cast<const float4*>(p);
  float4 f1 = *reinterpret_cast<const float4*>(p + 4);
  float4 f2 = *reinterpret_cast<const float4*>(p + 16);
  float4 f3 = *reinterpret_cast<const float4*>(p + 20);
  Bits256 b = {{pack2(f0.x, f0.y), pack2(f0.z, f0.w), pack2(f1.x, f1.y), pack2(f1.z, f1.w),
                pack2(f2.x, f2.y), pack2(f2.z, f2.w), pack2(f3.x, f3.y), pack2(f3.z, f3.w)}};
  return __builtin_bit_cast(v16bf, b);
}
// B (32x16 bf16): lane L holds column (L&15); k = (e>>3)*16 + ((L>>4)*8) + (e&7) (same as A).
// B[k][n] = W[colbase+n][kbase+k] with W row-major [O][ld] (GEMM is x @ W^T).
// Works for global or LDS-resident W (addrspace inferred after inlining -> ds_load_b128).
__device__ __forceinline__ v16bf load_b_bf16(const uint16_t* W, int ld,
                                             int colbase, int kbase, int lane) {
  int n = colbase + (lane & 15);
  int k = kbase + ((lane >> 4) << 3);
  const uint16_t* p = W + (size_t)n * ld + k;
  uint4 lo = *reinterpret_cast<const uint4*>(p);
  uint4 hi = *reinterpret_cast<const uint4*>(p + 16);
  Bits256 b = {{lo.x, lo.y, lo.z, lo.w, hi.x, hi.y, hi.z, hi.w}};
  return __builtin_bit_cast(v16bf, b);
}

#define WMMA_BF16(a, b, c) \
  __builtin_amdgcn_wmma_f32_16x16x32_bf16(false, (a), false, (b), (short)0, (c), false, false)

// cooperative global->LDS stage of a bf16 weight matrix (16B chunks)
__device__ __forceinline__ void stage_lds(uint16_t* sdst, const uint16_t* gsrc, int elems) {
  uint4* d16 = reinterpret_cast<uint4*>(sdst);
  const uint4* s16 = reinterpret_cast<const uint4*>(gsrc);
  for (int i = threadIdx.x; i < elems / 8; i += blockDim.x) d16[i] = s16[i];
  __syncthreads();
}

// ---------------- kernels ----------------

__global__ void cvt_bf16_kernel(const float* __restrict__ s, uint16_t* __restrict__ d, int n) {
  int i = blockIdx.x * blockDim.x + threadIdx.x;
  if (i < n) d[i] = f2bf(s[i]);
}

// h1 = lrelu(atom @ W1^T + b1); also h1 bf16 copy and hscore[n] = h1[n] . Ww[0:128]
// W1 (32KB bf16) staged in LDS; B fragments via ds_load_b128.
__global__ void node1_kernel(const uint16_t* __restrict__ atombf,
                             const uint16_t* __restrict__ w1bf,
                             const float* __restrict__ b1, const float* __restrict__ ww,
                             float* __restrict__ h1f, uint16_t* __restrict__ h1bf,
                             float* __restrict__ hscore, int ntiles) {
  __shared__ uint16_t sW[OUT_F * IN_F];
  stage_lds(sW, w1bf, OUT_F * IN_F);

  int lane = threadIdx.x & 31;
  int tile = blockIdx.x * (blockDim.x >> 5) + (threadIdx.x >> 5);
  if (tile >= ntiles) return;
  int rbase = tile * 16;
  int hi = lane >> 4, colc = lane & 15;
  v16bf a[4];
#pragma unroll
  for (int kt = 0; kt < 4; ++kt)
    a[kt] = load_a_bf16(atombf, (size_t)(rbase + colc), IN_F, kt * 32, lane);
  float hp[8] = {0, 0, 0, 0, 0, 0, 0, 0};
#pragma unroll
  for (int jn = 0; jn < 8; ++jn) {
    v8f acc = {};
#pragma unroll
    for (int kt = 0; kt < 4; ++kt)
      acc = WMMA_BF16(a[kt], load_b_bf16(sW, IN_F, jn * 16, kt * 32, lane), acc);
    int col = jn * 16 + colc;
    float bias = b1[col], w0 = ww[col];
#pragma unroll
    for (int r = 0; r < 8; ++r) {
      int row = rbase + r + (hi << 3);
      float v = lrelu(acc[r] + bias);
      h1f[(size_t)row * OUT_F + col] = v;
      h1bf[(size_t)row * OUT_F + col] = f2bf(v);
      hp[r] += v * w0;
    }
  }
#pragma unroll
  for (int r = 0; r < 8; ++r)
#pragma unroll
    for (int m = 1; m < 16; m <<= 1) hp[r] += __shfl_xor(hp[r], m, 32);
  if (colc == 0)
#pragma unroll
    for (int r = 0; r < 8; ++r) hscore[rbase + r + (hi << 3)] = hp[r];
}

// nt = h1 @ Wa^T + ba  (stored bf16; consumed only by the f32-accum gather)
// Wa (32KB bf16) staged in LDS.
__global__ void node2_kernel(const uint16_t* __restrict__ h1bf, const uint16_t* __restrict__ wabf,
                             const float* __restrict__ ba, uint16_t* __restrict__ ntbf, int ntiles) {
  __shared__ uint16_t sW[OUT_F * OUT_F];
  stage_lds(sW, wabf, OUT_F * OUT_F);

  int lane = threadIdx.x & 31;
  int tile = blockIdx.x * (blockDim.x >> 5) + (threadIdx.x >> 5);
  if (tile >= ntiles) return;
  int rbase = tile * 16;
  int hi = lane >> 4, colc = lane & 15;
  v16bf a[4];
#pragma unroll
  for (int kt = 0; kt < 4; ++kt)
    a[kt] = load_a_bf16(h1bf, (size_t)(rbase + colc), OUT_F, kt * 32, lane);
#pragma unroll
  for (int jn = 0; jn < 8; ++jn) {
    v8f acc = {};
#pragma unroll
    for (int kt = 0; kt < 4; ++kt)
      acc = WMMA_BF16(a[kt], load_b_bf16(sW, OUT_F, jn * 16, kt * 32, lane), acc);
    int col = jn * 16 + colc;
    float bias = ba[col];
#pragma unroll
    for (int r = 0; r < 8; ++r) {
      int row = rbase + r + (hi << 3);
      ntbf[(size_t)row * OUT_F + col] = f2bf(acc[r] + bias);
    }
  }
}

// Per 16-edge tile: nn_e = lrelu([atomBF[src]||bond] @ W2^T + b2), fused reduce vs Ww[128:256],
// emit logits[e] = lrelu(hscore[dst]+escore+bw). nn_e never touches memory.
// W2 (48KB bf16) staged in LDS once per block; B fragments come from ds_load_b128.
#define W2_ELEMS (OUT_F * (IN_F + BOND_F))   // 24576 ushorts = 48KB
__global__ void edge_kernel(const uint16_t* __restrict__ atombf, const float* __restrict__ bond,
                            const int* __restrict__ src, const int* __restrict__ dstv,
                            const uint16_t* __restrict__ w2bf, const float* __restrict__ b2,
                            const float* __restrict__ ww, const float* __restrict__ bwp,
                            const float* __restrict__ hscore, float* __restrict__ logits,
                            int etiles) {
  __shared__ uint16_t sW2[W2_ELEMS];
  stage_lds(sW2, w2bf, W2_ELEMS);

  int lane = threadIdx.x & 31;
  int tile = blockIdx.x * (blockDim.x >> 5) + (threadIdx.x >> 5);
  if (tile >= etiles) return;
  int ebase = tile * 16;
  int hi = lane >> 4, colc = lane & 15;
  int erow = ebase + colc;
  int srow = src[erow];
  if (tile + 1 < etiles)  // linear bond stream: hint next tile into cache
    __builtin_prefetch(bond + (size_t)(erow + 16) * BOND_F, 0, 1);
  v16bf a[6];
#pragma unroll
  for (int kt = 0; kt < 4; ++kt)
    a[kt] = load_a_bf16(atombf, (size_t)srow, IN_F, kt * 32, lane);  // L2-resident gather
  a[4] = load_a_f32(bond, (size_t)erow, BOND_F, 0, lane);
  a[5] = load_a_f32(bond, (size_t)erow, BOND_F, 32, lane);
  float ep[8] = {0, 0, 0, 0, 0, 0, 0, 0};
#pragma unroll
  for (int jn = 0; jn < 8; ++jn) {
    v8f acc = {};
#pragma unroll
    for (int kt = 0; kt < 6; ++kt)
      acc = WMMA_BF16(a[kt], load_b_bf16(sW2, IN_F + BOND_F, jn * 16, kt * 32, lane), acc);
    int col = jn * 16 + colc;
    float bias = b2[col], w1 = ww[OUT_F + col];
#pragma unroll
    for (int r = 0; r < 8; ++r) ep[r] += lrelu(acc[r] + bias) * w1;
  }
#pragma unroll
  for (int r = 0; r < 8; ++r)
#pragma unroll
    for (int m = 1; m < 16; m <<= 1) ep[r] += __shfl_xor(ep[r], m, 32);
  if (colc == 0) {
    float bwv = bwp[0];
#pragma unroll
    for (int r = 0; r < 8; ++r) {
      int e = ebase + r + (hi << 3);
      logits[e] = lrelu(ep[r] + hscore[dstv[e]] + bwv);
    }
  }
}

// Segment softmax (8 contiguous edges per node) + weighted nt[src] gather + ELU -> context bf16
__global__ void agg_kernel(const float* __restrict__ logits, const int* __restrict__ src,
                           const uint16_t* __restrict__ ntbf, uint16_t* __restrict__ ctxbf,
                           int nnodes) {
  int lane = threadIdx.x & 31;
  int n = blockIdx.x * (blockDim.x >> 5) + (threadIdx.x >> 5);
  if (n >= nnodes) return;
  float lg[DEG];
  int sr[DEG];
#pragma unroll
  for (int i = 0; i < DEG; ++i) {
    lg[i] = logits[(size_t)n * DEG + i];
    sr[i] = src[(size_t)n * DEG + i];
  }
  float m = lg[0];
#pragma unroll
  for (int i = 1; i < DEG; ++i) m = fmaxf(m, lg[i]);
  float s = 0.f, ex[DEG];
#pragma unroll
  for (int i = 0; i < DEG; ++i) { ex[i] = __expf(lg[i] - m); s += ex[i]; }
  float inv = 1.f / s;
  float acc[4] = {0, 0, 0, 0};
#pragma unroll
  for (int i = 0; i < DEG; ++i) {
    float w = ex[i] * inv;
    const uint16_t* row = ntbf + (size_t)sr[i] * OUT_F;
#pragma unroll
    for (int t = 0; t < 4; ++t) acc[t] += w * bf2f(row[lane + 32 * t]);
  }
#pragma unroll
  for (int t = 0; t < 4; ++t) {
    float x = acc[t];
    float c = x > 0.f ? x : (__expf(x) - 1.f);  // ELU, alpha=1
    ctxbf[(size_t)n * OUT_F + lane + 32 * t] = f2bf(c);
  }
}

// Fused GRU: gi = ctx @ W_ih^T + b_ih, gh = h1 @ W_hh^T + b_hh, gates, final.
// (W_ih+W_hh = 192KB combined: left in global/L2 to stay within safe static-LDS budget.)
__global__ void gru_kernel(const uint16_t* __restrict__ ctxbf, const uint16_t* __restrict__ h1bf,
                           const float* __restrict__ h1f, const uint16_t* __restrict__ wihbf,
                           const uint16_t* __restrict__ whhbf, const float* __restrict__ bih,
                           const float* __restrict__ bhh, float* __restrict__ finalv, int ntiles) {
  int lane = threadIdx.x & 31;
  int tile = blockIdx.x * (blockDim.x >> 5) + (threadIdx.x >> 5);
  if (tile >= ntiles) return;
  int rbase = tile * 16;
  int hi = lane >> 4, colc = lane & 15;
  v16bf ax[4], ah[4];
#pragma unroll
  for (int kt = 0; kt < 4; ++kt) {
    ax[kt] = load_a_bf16(ctxbf, (size_t)(rbase + colc), OUT_F, kt * 32, lane);
    ah[kt] = load_a_bf16(h1bf, (size_t)(rbase + colc), OUT_F, kt * 32, lane);
  }
#pragma unroll
  for (int jc = 0; jc < 8; ++jc) {
    v8f aIr = {}, aIz = {}, aIn = {}, aHr = {}, aHz = {}, aHn = {};
#pragma unroll
    for (int kt = 0; kt < 4; ++kt) {
      aIr = WMMA_BF16(ax[kt], load_b_bf16(wihbf, OUT_F, 0 * OUT_F + jc * 16, kt * 32, lane), aIr);
      aIz = WMMA_BF16(ax[kt], load_b_bf16(wihbf, OUT_F, 1 * OUT_F + jc * 16, kt * 32, lane), aIz);
      aIn = WMMA_BF16(ax[kt], load_b_bf16(wihbf, OUT_F, 2 * OUT_F + jc * 16, kt * 32, lane), aIn);
      aHr = WMMA_BF16(ah[kt], load_b_bf16(whhbf, OUT_F, 0 * OUT_F + jc * 16, kt * 32, lane), aHr);
      aHz = WMMA_BF16(ah[kt], load_b_bf16(whhbf, OUT_F, 1 * OUT_F + jc * 16, kt * 32, lane), aHz);
      aHn = WMMA_BF16(ah[kt], load_b_bf16(whhbf, OUT_F, 2 * OUT_F + jc * 16, kt * 32, lane), aHn);
    }
    int col = jc * 16 + colc;
    float bir = bih[col], biz = bih[OUT_F + col], bin_ = bih[2 * OUT_F + col];
    float bhr = bhh[col], bhz = bhh[OUT_F + col], bhn = bhh[2 * OUT_F + col];
#pragma unroll
    for (int r = 0; r < 8; ++r) {
      int row = rbase + r + (hi << 3);
      float rg = 1.f / (1.f + __expf(-((aIr[r] + bir) + (aHr[r] + bhr))));
      float zg = 1.f / (1.f + __expf(-((aIz[r] + biz) + (aHz[r] + bhz))));
      float ng = tanhf((aIn[r] + bin_) + rg * (aHn[r] + bhn));
      float h = h1f[(size_t)row * OUT_F + col];
      finalv[(size_t)row * OUT_F + col] = (1.f - zg) * ng + zg * h;
    }
  }
}

// ---------------- host launch ----------------

extern "C" void kernel_launch(void* const* d_in, const int* in_sizes, int n_in,
                              void* d_out, int out_size, void* d_ws, size_t ws_size,
                              hipStream_t stream) {
  const float* atom = (const float*)d_in[0];
  const float* bond = (const float*)d_in[1];
  const int* src = (const int*)d_in[2];
  const int* dst = (const int*)d_in[3];
  const float* W1 = (const float*)d_in[4];
  const float* b1 = (const float*)d_in[5];
  const float* W2 = (const float*)d_in[6];
  const float* b2 = (const float*)d_in[7];
  const float* Wa = (const float*)d_in[8];
  const float* ba = (const float*)d_in[9];
  const float* Ww = (const float*)d_in[10];
  const float* bw = (const float*)d_in[11];
  const float* W_ih = (const float*)d_in[12];
  const float* b_ih = (const float*)d_in[13];
  const float* W_hh = (const float*)d_in[14];
  const float* b_hh = (const float*)d_in[15];

  const int N = in_sizes[0] / IN_F;
  const int E = in_sizes[1] / BOND_F;
  const int NT = N / 16;   // 6250 full tiles (N=100000)
  const int ET = E / 16;   // 50000 full tiles (E=800000)

  float* finalv = (float*)d_out;
  float* h1f = finalv + (size_t)N * OUT_F;  // second output (h1), also reused by GRU

  // workspace carve-out (256B aligned): ~107 MB total
  char* base = (char*)d_ws;
  size_t off = 0;
  auto carve = [&](size_t bytes) {
    void* p = base + off;
    off = (off + bytes + 255) & ~(size_t)255;
    return p;
  };
  uint16_t* atombf = (uint16_t*)carve((size_t)N * IN_F * 2);
  uint16_t* h1bf = (uint16_t*)carve((size_t)N * OUT_F * 2);
  uint16_t* ntbf = (uint16_t*)carve((size_t)N * OUT_F * 2);
  uint16_t* ctxbf = (uint16_t*)carve((size_t)N * OUT_F * 2);
  float* hscore = (float*)carve((size_t)N * 4);
  float* logits = (float*)carve((size_t)E * 4);
  uint16_t* w1bf = (uint16_t*)carve((size_t)OUT_F * IN_F * 2);
  uint16_t* w2bf = (uint16_t*)carve((size_t)OUT_F * (IN_F + BOND_F) * 2);
  uint16_t* wabf = (uint16_t*)carve((size_t)OUT_F * OUT_F * 2);
  uint16_t* wihbf = (uint16_t*)carve((size_t)3 * OUT_F * OUT_F * 2);
  uint16_t* whhbf = (uint16_t*)carve((size_t)3 * OUT_F * OUT_F * 2);
  (void)ws_size; (void)n_in; (void)out_size;

  auto cvt = [&](const float* s, uint16_t* d, int n) {
    cvt_bf16_kernel<<<(n + 255) / 256, 256, 0, stream>>>(s, d, n);
  };
  cvt(atom, atombf, N * IN_F);
  cvt(W1, w1bf, OUT_F * IN_F);
  cvt(W2, w2bf, OUT_F * (IN_F + BOND_F));
  cvt(Wa, wabf, OUT_F * OUT_F);
  cvt(W_ih, wihbf, 3 * OUT_F * OUT_F);
  cvt(W_hh, whhbf, 3 * OUT_F * OUT_F);

  node1_kernel<<<(NT + 7) / 8, 256, 0, stream>>>(atombf, w1bf, b1, Ww, h1f, h1bf, hscore, NT);
  node2_kernel<<<(NT + 7) / 8, 256, 0, stream>>>(h1bf, wabf, ba, ntbf, NT);
  edge_kernel<<<(ET + 7) / 8, 256, 0, stream>>>(atombf, bond, src, dst, w2bf, b2, Ww, bw,
                                                hscore, logits, ET);
  agg_kernel<<<(N + 7) / 8, 256, 0, stream>>>(logits, src, ntbf, ctxbf, N);
  gru_kernel<<<(NT + 7) / 8, 256, 0, stream>>>(ctxbf, h1bf, h1f, wihbf, whhbf, b_ih, b_hh,
                                               finalv, NT);
}
